// ComplexDotProductAttention_13374528160386
// MI455X (gfx1250) — matrix operand order repaired and checked by reading, not compiled, not run on hardware
//
#include <hip/hip_runtime.h>

// ---------- types ----------
typedef __bf16 bf16_t;
typedef __attribute__((ext_vector_type(16))) __bf16 v16bf;
typedef __attribute__((ext_vector_type(8)))  __bf16 v8bf;
typedef __attribute__((ext_vector_type(2)))  __bf16 v2bf;
typedef __attribute__((ext_vector_type(16))) unsigned short v16u;
typedef __attribute__((ext_vector_type(8)))  float  v8f;
typedef __attribute__((ext_vector_type(4)))  float  v4f;
typedef __attribute__((ext_vector_type(2)))  float  v2f;

#define WMMA_BF16(A, B, C) \
  __builtin_amdgcn_wmma_f32_16x16x32_bf16(false, (A), false, (B), (short)0, (C), false, false)

// async global->LDS copy, GVS mode: mem = SADDR(64) + VADDR(32); LDS = VDST(32)
#define ASYNC_B128(ldsaddr, voff, sbase) \
  asm volatile("global_load_async_to_lds_b128 %0, %1, %2" \
               :: "v"(ldsaddr), "v"(voff), "s"(sbase) : "memory")
#define WAIT_ASYNC0() asm volatile("s_wait_asynccnt 0x0" ::: "memory")

// ---------- problem constants ----------
constexpr int Bb = 16, Qq = 2048, Kk = 2048, Dd = 64, Vv = 64;
constexpr int KT    = 64;   // keys per iteration
constexpr int QROWS = 64;   // query rows per workgroup (4 rowgroups x 16)
constexpr float SCALE_LOG2E = 0.18033688011112042f; // (1/sqrt(64)) * log2(e)

// ---------- LDS layout (byte offsets; all 16B aligned) ----------
constexpr int K_STRIDE  = 136; // bf16 elems (128 + 8 pad) -> 272B rows
constexpr int QC_STRIDE = 136;
constexpr int VT_STRIDE = 72;  // bf16 elems (64 + 8 pad)  -> 144B rows
constexpr int P_STRIDE  = 72;
constexpr int OFF_K  = 0;                    // 64*136*2  = 17408   Kcat = [kr | -ki]
constexpr int OFF_VT = 17408;                // 128*72*2  = 18432   [vr|vi]^T
constexpr int OFF_P  = 35840;                // 8*16*72*2 = 18432   per-wave P scratch
constexpr int SMEM_BYTES = 54272;
// overlays: Q tile (17408B) sits in Vt region during prologue;
//           epilogue f32 combine buffer (32768B) sits over dead K/Vt.

union AB { v16bf v; v8bf h[2]; };

__device__ __forceinline__ v2bf pk2(float a, float b) {
  v2bf t; t[0] = (bf16_t)a; t[1] = (bf16_t)b; return t;
}

// DPP16 butterfly step (pure VALU, stays within 16-lane row halves)
template<int CTRL>
__device__ __forceinline__ float dppf(float x) {
  return __builtin_bit_cast(float,
      __builtin_amdgcn_update_dpp(0, __builtin_bit_cast(int, x), CTRL, 0xf, 0xf, true));
}
__device__ __forceinline__ float rowmax16(float x) {
  x = fmaxf(x, dppf<0xB1>(x));   // quad_perm [1,0,3,2]
  x = fmaxf(x, dppf<0x4E>(x));   // quad_perm [2,3,0,1]
  x = fmaxf(x, dppf<0x141>(x));  // ROW_HALF_MIRROR
  x = fmaxf(x, dppf<0x140>(x));  // ROW_MIRROR
  return x;
}
__device__ __forceinline__ float rowsum16(float x) {
  x += dppf<0xB1>(x);
  x += dppf<0x4E>(x);
  x += dppf<0x141>(x);
  x += dppf<0x140>(x);
  return x;
}

// ===================== preprocessing (one-shot, memory-bound) =================
// [B,N,64,2] f32 -> [B,N,128] bf16 rows: [re | (negIm? -im : im)]
__global__ __launch_bounds__(256) void prep_qk_kernel(
    const float* __restrict__ src, bf16_t* __restrict__ dst, int negIm) {
  int t   = blockIdx.x * 256 + threadIdx.x;  // B*N*16 threads
  int row = t >> 4, g = t & 15;
  const float* p = src + ((size_t)row * Dd + 4 * g) * 2;
  v4f x = *(const v4f*)p;
  v4f y = *(const v4f*)(p + 4);
  float s = negIm ? -1.f : 1.f;
  bf16_t* d = dst + (size_t)row * 128 + 4 * g;
  *(v2bf*)(d)      = pk2(x.x, x.z);
  *(v2bf*)(d + 2)  = pk2(y.x, y.z);
  *(v2bf*)(d + 64) = pk2(s * x.y, s * x.w);
  *(v2bf*)(d + 66) = pk2(s * y.y, s * y.w);
}

// values [B,K,64,2] f32 -> Vt [B,128,K] bf16 (vc<64: vr, vc>=64: vi), kk-major
__global__ __launch_bounds__(256) void prep_v_kernel(
    const float* __restrict__ v, bf16_t* __restrict__ dst) {
  int t  = blockIdx.x * 256 + threadIdx.x;   // B*(K/2)*16 threads
  int g  = t & 15;
  int pr = (t >> 4) & (Kk / 2 - 1);
  int b  = t >> 14;
  const float* p0 = v + (((size_t)b * Kk + 2 * pr) * Vv + 4 * g) * 2;
  const float* p1 = p0 + Vv * 2;
  v4f x0 = *(const v4f*)p0, y0 = *(const v4f*)(p0 + 4);
  v4f x1 = *(const v4f*)p1, y1 = *(const v4f*)(p1 + 4);
  bf16_t* base = dst + (size_t)b * 128 * Kk;
  int col = 2 * pr;
  *(v2bf*)(base + (size_t)(4 * g + 0) * Kk + col) = pk2(x0.x, x1.x);
  *(v2bf*)(base + (size_t)(4 * g + 1) * Kk + col) = pk2(x0.z, x1.z);
  *(v2bf*)(base + (size_t)(4 * g + 2) * Kk + col) = pk2(y0.x, y1.x);
  *(v2bf*)(base + (size_t)(4 * g + 3) * Kk + col) = pk2(y0.z, y1.z);
  bf16_t* basei = base + (size_t)64 * Kk;
  *(v2bf*)(basei + (size_t)(4 * g + 0) * Kk + col) = pk2(x0.y, x1.y);
  *(v2bf*)(basei + (size_t)(4 * g + 1) * Kk + col) = pk2(x0.w, x1.w);
  *(v2bf*)(basei + (size_t)(4 * g + 2) * Kk + col) = pk2(y0.y, y1.y);
  *(v2bf*)(basei + (size_t)(4 * g + 3) * Kk + col) = pk2(y0.w, y1.w);
}

// ========================= fused flash attention ==============================
__global__ __launch_bounds__(256) void cattn_wmma_kernel(
    const bf16_t* __restrict__ qg, const bf16_t* __restrict__ kg,
    const bf16_t* __restrict__ vg, float* __restrict__ out)
{
  __shared__ __align__(16) char smem[SMEM_BYTES];
  bf16_t* sK  = (bf16_t*)(smem + OFF_K);
  bf16_t* sVt = (bf16_t*)(smem + OFF_VT);
  bf16_t* sP  = (bf16_t*)(smem + OFF_P);
  bf16_t* sQ  = (bf16_t*)(smem + OFF_VT);  // prologue-only overlay
  float*  sC  = (float*)smem;              // epilogue-only overlay

  const int tid  = threadIdx.x;
  const int wave = tid >> 5;
  const int lane = tid & 31;
  const int rg   = wave >> 1;
  const int head = wave & 1;                 // 0: real head, 1: imag head
  const int b    = blockIdx.x >> 5;
  const int q0   = (blockIdx.x & 31) * QROWS;
  const int ln   = lane & 15;
  const int hi   = lane >> 4;

  const uint32_t sbase = (uint32_t)(size_t)(&smem[0]);  // LDS byte offset of smem

  // ---------- async-stage Q tile (bf16, pre-converted) into sQ overlay -------
  {
    uint64_t qb = (uint64_t)(const void*)(qg + ((size_t)b * Qq + q0) * 128);
    #pragma unroll
    for (int i = 0; i < 4; ++i) {
      int u = tid + 256 * i;                 // row=u>>4 (0..63), seg=u&15
      uint32_t lds  = sbase + (uint32_t)(OFF_VT + (u >> 4) * (QC_STRIDE * 2) + (u & 15) * 16);
      uint32_t voff = (uint32_t)(((u >> 4) * 128 + (u & 15) * 8) * 2);
      ASYNC_B128(lds, voff, qb);
    }
    WAIT_ASYNC0();
  }
  __syncthreads();

  // == per-wave Q A-fragments in registers. head0: [qr|qi]; head1: [qi|-qr] ===
  AB Afrag[4];
  {
    const bf16_t* qrow = sQ + (rg * 16 + ln) * QC_STRIDE;
    #pragma unroll
    for (int c = 0; c < 4; ++c) {
      int src  = head ? (((c + 2) & 3) * 32) : (c * 32);
      int base = src + hi * 8;
      Afrag[c].h[0] = *(const v8bf*)(qrow + base);
      Afrag[c].h[1] = *(const v8bf*)(qrow + base + 16);
      if (head && c >= 2) {                  // negate qr chunks (sign-bit xor)
        v16u u16 = __builtin_bit_cast(v16u, Afrag[c].v);
        u16 = u16 ^ (v16u)((unsigned short)0x8000);
        Afrag[c].v = __builtin_bit_cast(v16bf, u16);
      }
    }
  }

  v8f acc[8];
  #pragma unroll
  for (int i = 0; i < 8; ++i) acc[i] = v8f{};
  float m[8], l[8];
  #pragma unroll
  for (int r = 0; r < 8; ++r) { m[r] = -__builtin_inff(); l[r] = 0.f; }

  auto loadKB = [&](AB& Bf, int nt, int c) {
    const bf16_t* kp = sK + (nt * 16 + ln) * K_STRIDE + c * 32 + hi * 8;
    Bf.h[0] = *(const v8bf*)kp;
    Bf.h[1] = *(const v8bf*)(kp + 16);
  };
  auto loadVB = [&](AB& Bf, int vnt, int c) {
    const bf16_t* vp = sVt + (vnt * 16 + ln) * VT_STRIDE + c * 32 + hi * 8;
    Bf.h[0] = *(const v8bf*)vp;
    Bf.h[1] = *(const v8bf*)(vp + 16);
  };

  // per-thread constant LDS targets + running global voffsets for staging
  const uint64_t kb = (uint64_t)(const void*)(kg + (size_t)b * Kk * 128);
  const uint64_t vb = (uint64_t)(const void*)(vg + (size_t)b * 128 * Kk);
  uint32_t klds[4], vlds[4], kgo[4], vgo[4];
  #pragma unroll
  for (int i = 0; i < 4; ++i) {
    int u = tid + 256 * i;
    klds[i] = sbase + (uint32_t)(OFF_K + (u >> 4) * (K_STRIDE * 2) + (u & 15) * 16);
    kgo[i]  = (uint32_t)(((u >> 4) * 128 + (u & 15) * 8) * 2);       // 64 rows x 256B
    vlds[i] = sbase + (uint32_t)(OFF_VT + (u >> 3) * (VT_STRIDE * 2) + (u & 7) * 16);
    vgo[i]  = (uint32_t)(((u >> 3) * Kk + (u & 7) * 8) * 2);         // 128 rows x 128B
  }

  // ======================== K loop (flash attention) =========================
  for (int k0 = 0; k0 < Kk; k0 += KT) {
    __syncthreads();                       // previous tile fully consumed

    // ---- pure async global->LDS staging (no VALU, no VGPR round-trip) ----
    #pragma unroll
    for (int i = 0; i < 4; ++i) ASYNC_B128(klds[i], kgo[i], kb);
    #pragma unroll
    for (int i = 0; i < 4; ++i) ASYNC_B128(vlds[i], vgo[i], vb);
    #pragma unroll
    for (int i = 0; i < 4; ++i) { kgo[i] += KT * 128 * 2; vgo[i] += KT * 2; }
    WAIT_ASYNC0();
    __syncthreads();

    // ---- scores: 4 N-tiles of 16 keys, K-dim 128 (16 WMMAs) ----
    v8f S[4];
    #pragma unroll
    for (int nt = 0; nt < 4; ++nt) S[nt] = v8f{};
    {
      AB Bf[2];
      loadKB(Bf[0], 0, 0);
      #pragma unroll
      for (int i = 0; i < 16; ++i) {
        int nt = i >> 2, c = i & 3;
        if (i + 1 < 16) loadKB(Bf[(i + 1) & 1], (i + 1) >> 2, (i + 1) & 3);
        S[nt] = WMMA_BF16(Afrag[c].v, Bf[i & 1].v, S[nt]);
      }
    }

    // ---- online softmax over 64 keys (base-2 domain; DPP16 reductions) ----
    float cf[8];
    #pragma unroll
    for (int r = 0; r < 8; ++r) {
      float t0 = S[0][r] * SCALE_LOG2E;
      float t1 = S[1][r] * SCALE_LOG2E;
      float t2 = S[2][r] * SCALE_LOG2E;
      float t3 = S[3][r] * SCALE_LOG2E;
      float mx = rowmax16(fmaxf(fmaxf(t0, t1), fmaxf(t2, t3)));
      float mn = fmaxf(m[r], mx);
      cf[r] = __builtin_amdgcn_exp2f(m[r] - mn);
      m[r]  = mn;
      float p0 = __builtin_amdgcn_exp2f(t0 - mn);
      float p1 = __builtin_amdgcn_exp2f(t1 - mn);
      float p2 = __builtin_amdgcn_exp2f(t2 - mn);
      float p3 = __builtin_amdgcn_exp2f(t3 - mn);
      S[0][r] = p0; S[1][r] = p1; S[2][r] = p2; S[3][r] = p3;
      l[r] = l[r] * cf[r] + rowsum16((p0 + p1) + (p2 + p3));
    }
    #pragma unroll
    for (int nt = 0; nt < 8; ++nt)
      #pragma unroll
      for (int r = 0; r < 8; ++r)
        acc[nt][r] *= cf[r];

    // ---- P: C-layout f32 -> A-layout bf16 via per-wave LDS scratch ----
    bf16_t* pscr = sP + wave * 16 * P_STRIDE;
    #pragma unroll
    for (int nt = 0; nt < 4; ++nt)
      #pragma unroll
      for (int r = 0; r < 8; ++r)
        pscr[(r + 8 * hi) * P_STRIDE + nt * 16 + ln] = (bf16_t)S[nt][r];
    asm volatile("" ::: "memory");         // keep DS store->load order

    // ---- O += P @ [vr|vi]: 2 K-chunks x 8 N-tiles (16 WMMAs) ----
    #pragma unroll
    for (int c = 0; c < 2; ++c) {
      AB Pf;
      const bf16_t* prow = pscr + ln * P_STRIDE + c * 32 + hi * 8;
      Pf.h[0] = *(const v8bf*)prow;
      Pf.h[1] = *(const v8bf*)(prow + 16);
      AB Bv[2];
      loadVB(Bv[0], 0, c);
      #pragma unroll
      for (int vnt = 0; vnt < 8; ++vnt) {
        if (vnt + 1 < 8) loadVB(Bv[(vnt + 1) & 1], vnt + 1, c);
        acc[vnt] = WMMA_BF16(Pf.v, Bv[vnt & 1].v, acc[vnt]);
      }
    }
  }

  // ======================== epilogue: normalize & combine =====================
  float rl[8];
  #pragma unroll
  for (int r = 0; r < 8; ++r) rl[r] = __builtin_amdgcn_rcpf(l[r]);
  #pragma unroll
  for (int nt = 0; nt < 8; ++nt)
    #pragma unroll
    for (int r = 0; r < 8; ++r)
      acc[nt][r] *= rl[r];

  __syncthreads();                          // K/Vt LDS dead; safe to overlay
  if (head == 1) {                          // imag head publishes [ir | ii]
    float* dst = sC + (rg * 16) * 128;
    #pragma unroll
    for (int nt = 0; nt < 8; ++nt)
      #pragma unroll
      for (int r = 0; r < 8; ++r)
        dst[(r + 8 * hi) * 128 + nt * 16 + ln] = acc[nt][r];
  }
  __syncthreads();
  if (head == 0) {                          // real head combines & stores
    const float* srcB = sC + (rg * 16) * 128;
    #pragma unroll
    for (int nt = 0; nt < 4; ++nt) {
      #pragma unroll
      for (int r = 0; r < 8; ++r) {
        int row = r + 8 * hi;
        int vc  = nt * 16 + ln;
        float ir = srcB[row * 128 + vc];
        float ii = srcB[row * 128 + 64 + vc];
        float yr = acc[nt][r]     - ii;
        float yi = acc[nt + 4][r] + ir;
        v2f o2; o2[0] = yr; o2[1] = yi;
        float* o = out + (((size_t)b * Qq + q0 + rg * 16 + row) * Vv + vc) * 2;
        *(v2f*)o = o2;
      }
    }
  }
}

extern "C" void kernel_launch(void* const* d_in, const int* in_sizes, int n_in,
                              void* d_out, int out_size, void* d_ws, size_t ws_size,
                              hipStream_t stream) {
  const float* qf = (const float*)d_in[0];
  const float* kf = (const float*)d_in[1];
  const float* vf = (const float*)d_in[2];
  float* out = (float*)d_out;

  // workspace: bf16 Qcat / Kcat / Vt  (3 x 8 MB)
  bf16_t* qg = (bf16_t*)d_ws;
  bf16_t* kg = qg + (size_t)Bb * Qq * 128;
  bf16_t* vg = kg + (size_t)Bb * Kk * 128;

  prep_qk_kernel<<<dim3(Bb * Qq * 16 / 256), dim3(256), 0, stream>>>(qf, qg, 0);
  prep_qk_kernel<<<dim3(Bb * Kk * 16 / 256), dim3(256), 0, stream>>>(kf, kg, 1);
  prep_v_kernel <<<dim3(Bb * (Kk / 2) * 16 / 256), dim3(256), 0, stream>>>(vf, vg);

  cattn_wmma_kernel<<<dim3(Bb * (Qq / QROWS)), dim3(256), 0, stream>>>(qg, kg, vg, out);
}
